// LatentAttentionBlock_21199958573286
// MI455X (gfx1250) — matrix operand order repaired and checked by reading, not compiled, not run on hardware
//
#include <hip/hip_runtime.h>
#include <hip/hip_bf16.h>
#include <cstdint>
#include <cstddef>

// ---------------- problem constants ----------------
static constexpr int BB   = 32;
static constexpr int SS   = 2048;
static constexpr int DIM  = 384;
static constexpr int QK   = 48;
static constexpr int EXPD = 768;               // 2*DIM
static constexpr int ROWS = BB * SS;           // 65536
static constexpr float SCALE = 0.14433756729740643f; // 1/sqrt(48)

typedef __attribute__((ext_vector_type(16))) _Float16 v16h;
typedef __attribute__((ext_vector_type(8)))  float    v8f;

// LDS tile row strides (padded to spread banks, still 16B aligned)
static constexpr int XST = 40;  // for 32-wide K tiles
static constexpr int KST = 72;  // for 64-wide K tile (attention keys)

// ---------------- WMMA helpers (CDNA5 16x16x32 f16 -> f32) ----------------
__device__ __forceinline__ v8f wmma16(v16h a, v16h b, v8f c) {
  return __builtin_amdgcn_wmma_f32_16x16x32_f16(false, a, false, b, (short)0, c,
                                                false, false);
}

// A-operand (16x32 f16). Lane m=lane%16 reads row m. half=lane/16.
// a[0..15] = row[half*16 + 0..15]
__device__ __forceinline__ v16h loadA(const _Float16* rowp, int half) {
  union { v16h h; uint4 q[2]; } u;
  u.q[0] = *(const uint4*)(rowp + half * 16);
  u.q[1] = *(const uint4*)(rowp + half * 16 + 8);
  return u.h;
}

// B-operand: Bs[n][k] storage (k contiguous). b[i] = Bs[n][kb+i].
__device__ __forceinline__ v16h loadB(const _Float16* rowp, int kb) {
  union { v16h h; uint4 q[2]; } u;
  const uint4* p = (const uint4*)(rowp + kb);
  u.q[0] = p[0];
  u.q[1] = p[1];
  return u.h;
}

__device__ __forceinline__ uint32_t pack2h(float a, float b) {
  union { _Float16 h[2]; uint32_t u; } p;
  p.h[0] = (_Float16)a;
  p.h[1] = (_Float16)b;
  return p.u;
}

// ---------------- Kernel 1: LayerNorm -> f16 ----------------
__global__ __launch_bounds__(256) void k_norm(const float* __restrict__ x,
                                              const float* __restrict__ nw,
                                              _Float16* __restrict__ xn) {
  int wave = threadIdx.x >> 5;
  int lane = threadIdx.x & 31;
  size_t row = (size_t)blockIdx.x * 8 + wave;
  const float* xr = x + row * DIM;
  float vals[12];
  float s = 0.f, s2 = 0.f;
#pragma unroll
  for (int i = 0; i < 12; ++i) {
    float v = xr[lane + 32 * i];
    vals[i] = v;
    s += v;
    s2 += v * v;
  }
#pragma unroll
  for (int off = 16; off > 0; off >>= 1) {
    s  += __shfl_xor(s,  off, 32);
    s2 += __shfl_xor(s2, off, 32);
  }
  float mu   = s * (1.0f / DIM);
  float var  = s2 * (1.0f / DIM) - mu * mu;
  float rstd = rsqrtf(var + 1e-5f);
  _Float16* o = xn + row * DIM;
#pragma unroll
  for (int i = 0; i < 12; ++i) {
    int d = lane + 32 * i;
    o[d] = (_Float16)((vals[i] - mu) * rstd * nw[d]);
  }
}

// ---------------- Kernel 2a: expand GEMM, q/k columns (0..95) ----------------
__global__ __launch_bounds__(128) void k_expand_qk(const _Float16* __restrict__ Xn,
                                                   const float* __restrict__ ew,
                                                   _Float16* __restrict__ Qb,
                                                   _Float16* __restrict__ Kb) {
  __shared__ __align__(16) _Float16 Xs[64][XST];
  __shared__ __align__(16) _Float16 Bs[96][XST];
  int row0 = blockIdx.x * 64;
  int tid  = threadIdx.x;
  int wv   = tid >> 5, lane = tid & 31, half = lane >> 4, n16 = lane & 15;

  v8f C[6];
#pragma unroll
  for (int t = 0; t < 6; ++t)
#pragma unroll
    for (int v = 0; v < 8; ++v) C[t][v] = 0.f;

  for (int k0 = 0; k0 < DIM; k0 += 32) {
    __syncthreads();
    // Xs: 64 rows x 32 f16 = 256 uint4, 128 threads x 2
#pragma unroll
    for (int i = 0; i < 2; ++i) {
      int idx = tid + 128 * i;
      int r = idx >> 2, part = idx & 3;
      *(uint4*)&Xs[r][part * 8] =
          *(const uint4*)(Xn + (size_t)(row0 + r) * DIM + k0 + part * 8);
    }
    // Bs: 96 rows x 32 f32 -> f16, 768 float4, 128 threads x 6
#pragma unroll
    for (int i = 0; i < 6; ++i) {
      int idx = tid + 128 * i;
      int n = idx >> 3, kq = (idx & 7) * 4;
      float4 f = *(const float4*)(ew + (size_t)n * DIM + k0 + kq);
      uint2 u;
      u.x = pack2h(f.x, f.y);
      u.y = pack2h(f.z, f.w);
      *(uint2*)&Bs[n][kq] = u;
    }
    if (k0 + 32 < DIM) __builtin_prefetch(Xn + (size_t)(row0 + tid) * DIM + k0 + 32, 0, 1);
    __syncthreads();
    v16h a = loadA(&Xs[wv * 16 + n16][0], half);
#pragma unroll
    for (int nt = 0; nt < 6; ++nt) {
      v16h b = loadB(&Bs[nt * 16 + n16][0], half * 16);
      C[nt] = wmma16(a, b, C[nt]);
    }
  }
#pragma unroll
  for (int nt = 0; nt < 6; ++nt) {
#pragma unroll
    for (int v = 0; v < 8; ++v) {
      size_t row = (size_t)row0 + wv * 16 + v + 8 * half;
      int e = nt * 16 + n16;
      if (nt < 3) Qb[row * 64 + e] = (_Float16)C[nt][v];
      else        Kb[row * 64 + (e - 48)] = (_Float16)C[nt][v];
    }
  }
  // zero pad cols 48..63 of Q and K (one b128 per thread per buffer)
  {
    uint4 z; z.x = 0u; z.y = 0u; z.z = 0u; z.w = 0u;
    int r = tid >> 1, h8 = (tid & 1) * 8;
    *(uint4*)&Qb[(size_t)(row0 + r) * 64 + 48 + h8] = z;
    *(uint4*)&Kb[(size_t)(row0 + r) * 64 + 48 + h8] = z;
  }
}

// ------- Kernel 2b: expand GEMM lin+pre pair, fused exact-GeGLU -------
__global__ __launch_bounds__(128) void k_expand_geglu(const _Float16* __restrict__ Xn,
                                                      const float* __restrict__ ew,
                                                      _Float16* __restrict__ G) {
  __shared__ __align__(16) _Float16 Xs[64][XST];
  __shared__ __align__(16) _Float16 Bl[64][XST];
  __shared__ __align__(16) _Float16 Bp[64][XST];
  int row0 = blockIdx.x * 64;
  int g0   = blockIdx.y * 64;
  int tid  = threadIdx.x;
  int wv   = tid >> 5, lane = tid & 31, half = lane >> 4, n16 = lane & 15;

  v8f Cl[4], Cp[4];
#pragma unroll
  for (int t = 0; t < 4; ++t)
#pragma unroll
    for (int v = 0; v < 8; ++v) { Cl[t][v] = 0.f; Cp[t][v] = 0.f; }

  for (int k0 = 0; k0 < DIM; k0 += 32) {
    __syncthreads();
#pragma unroll
    for (int i = 0; i < 2; ++i) {
      int idx = tid + 128 * i;
      int r = idx >> 2, part = idx & 3;
      *(uint4*)&Xs[r][part * 8] =
          *(const uint4*)(Xn + (size_t)(row0 + r) * DIM + k0 + part * 8);
    }
#pragma unroll
    for (int i = 0; i < 4; ++i) {
      int idx = tid + 128 * i;
      int n = idx >> 3, kq = (idx & 7) * 4;
      float4 fl = *(const float4*)(ew + (size_t)(96 + g0 + n) * DIM + k0 + kq);
      float4 fp = *(const float4*)(ew + (size_t)(864 + g0 + n) * DIM + k0 + kq);
      uint2 ul, up;
      ul.x = pack2h(fl.x, fl.y); ul.y = pack2h(fl.z, fl.w);
      up.x = pack2h(fp.x, fp.y); up.y = pack2h(fp.z, fp.w);
      *(uint2*)&Bl[n][kq] = ul;
      *(uint2*)&Bp[n][kq] = up;
    }
    __syncthreads();
    v16h a = loadA(&Xs[wv * 16 + n16][0], half);
#pragma unroll
    for (int nt = 0; nt < 4; ++nt) {
      v16h bl = loadB(&Bl[nt * 16 + n16][0], half * 16);
      Cl[nt] = wmma16(a, bl, Cl[nt]);
      v16h bp = loadB(&Bp[nt * 16 + n16][0], half * 16);
      Cp[nt] = wmma16(a, bp, Cp[nt]);
    }
  }
#pragma unroll
  for (int nt = 0; nt < 4; ++nt) {
#pragma unroll
    for (int v = 0; v < 8; ++v) {
      size_t row = (size_t)row0 + wv * 16 + v + 8 * half;
      int g = g0 + nt * 16 + n16;
      float lin = Cl[nt][v];
      float pre = Cp[nt][v];
      float gel = 0.5f * pre * (1.0f + erff(pre * 0.70710678118654752f));
      G[row * (size_t)EXPD + g] = (_Float16)(lin * gel);
    }
  }
}

// ---------------- Kernel 3: causal flash attention ----------------
// grid (S/64, B), 256 threads (8 waves). Wave w: rows (w/2)*16, col-half w&1.
__global__ __launch_bounds__(256, 1) void k_attn(const _Float16* __restrict__ Qb,
                                                 const _Float16* __restrict__ Kb,
                                                 const _Float16* __restrict__ G,
                                                 const float* __restrict__ pm,
                                                 _Float16* __restrict__ AT) {
  __shared__ __align__(16) _Float16 Ks[32][KST];    // keys x dims (padded)
  __shared__ __align__(16) _Float16 VsT[384][XST];  // cols x keys (padded)
  __shared__ __align__(16) _Float16 Pst[8][16][XST];

  int b  = blockIdx.y;
  int qt = blockIdx.x;
  int tid = threadIdx.x;
  int wv = tid >> 5, lane = tid & 31, half = lane >> 4, n16 = lane & 15;
  int rg = wv >> 1, ch = wv & 1;

  float sp = log1pf(__expf(pm[0])); // softplus(pos_mult)

  size_t qrow0 = (size_t)b * SS + qt * 64 + rg * 16;

  // Q A-operands (K=64 -> two 16x32 chunks), loaded directly from global, contiguous k order (16*half + i)
  v16h aQ0 = loadA(Qb + (qrow0 + n16) * 64 + 0,  half);
  v16h aQ1 = loadA(Qb + (qrow0 + n16) * 64 + 32, half);

  float m_r[8], l_r[8];
  v8f O[12];
#pragma unroll
  for (int v = 0; v < 8; ++v) { m_r[v] = -1e30f; l_r[v] = 0.f; }
#pragma unroll
  for (int ct = 0; ct < 12; ++ct)
#pragma unroll
    for (int v = 0; v < 8; ++v) O[ct][v] = 0.f;

  int ntile = 2 * qt + 2;
  for (int jt = 0; jt < ntile; ++jt) {
    int jb = jt * 32;
    __syncthreads();
    // --- stage K tile (32 keys x 64 dims) via async global->LDS (ASYNCcnt)
    {
      int key = tid >> 3, part = tid & 7;
      uint32_t ldsoff = (uint32_t)(uintptr_t)&Ks[key][part * 8];
      const _Float16* gp = Kb + ((size_t)b * SS + jb + key) * 64 + part * 8;
      asm volatile("global_load_async_to_lds_b128 %0, %1, off"
                   :: "v"(ldsoff), "v"(gp) : "memory");
    }
    // --- stage V tile transposed: VsT[col][key] = G[row][384+col]
#pragma unroll
    for (int p = 0; p < 3; ++p) {
#pragma unroll
      for (int i = 0; i < 8; ++i) {
        int idx = tid + 256 * i;        // 0..2047
        int key = idx >> 6;             // 0..31
        int c   = p * 128 + (idx & 63) * 2;
        union { uint32_t u; _Float16 h[2]; } d;
        d.u = *(const uint32_t*)(G + ((size_t)b * SS + jb + key) * EXPD + 384 + c);
        VsT[c][key]     = d.h[0];
        VsT[c + 1][key] = d.h[1];
      }
    }
    asm volatile("s_wait_asynccnt 0x0" ::: "memory");
    __syncthreads();

    // --- scores: 16 rows x 32 keys
    v8f S0, S1;
#pragma unroll
    for (int v = 0; v < 8; ++v) { S0[v] = 0.f; S1[v] = 0.f; }
    S0 = wmma16(aQ0, loadB(&Ks[n16][0],      half * 16),      S0);
    S0 = wmma16(aQ1, loadB(&Ks[n16][0],      32 + half * 16), S0);
    S1 = wmma16(aQ0, loadB(&Ks[16 + n16][0], half * 16),      S1);
    S1 = wmma16(aQ1, loadB(&Ks[16 + n16][0], 32 + half * 16), S1);

    // --- online softmax with causal mask + linear positional bias
    float alpha_v[8], p0[8], p1[8];
    int iBase = qt * 64 + rg * 16 + 8 * half;
    int j0 = jb + n16, j1 = jb + 16 + n16;
#pragma unroll
    for (int v = 0; v < 8; ++v) {
      int i = iBase + v;
      float s0 = S0[v] * SCALE + sp * (float)(j0 - i);
      float s1 = S1[v] * SCALE + sp * (float)(j1 - i);
      s0 = (j0 > i) ? -1e30f : s0;
      s1 = (j1 > i) ? -1e30f : s1;
      float t = fmaxf(s0, s1);
#pragma unroll
      for (int off = 8; off > 0; off >>= 1) t = fmaxf(t, __shfl_xor(t, off, 32));
      float mn = fmaxf(m_r[v], t);
      float e0 = __expf(s0 - mn), e1 = __expf(s1 - mn);
      float rs = e0 + e1;
#pragma unroll
      for (int off = 8; off > 0; off >>= 1) rs += __shfl_xor(rs, off, 32);
      float al = __expf(m_r[v] - mn);
      l_r[v] = l_r[v] * al + rs;
      m_r[v] = mn;
      alpha_v[v] = al;
      p0[v] = e0;
      p1[v] = e1;
    }
#pragma unroll
    for (int ct = 0; ct < 12; ++ct)
#pragma unroll
      for (int v = 0; v < 8; ++v) O[ct][v] *= alpha_v[v];

    // --- stage P (C-layout -> LDS row-major), then read as A-operand
#pragma unroll
    for (int v = 0; v < 8; ++v) {
      int r = v + 8 * half;
      Pst[wv][r][n16]      = (_Float16)p0[v];
      Pst[wv][r][16 + n16] = (_Float16)p1[v];
    }
    v16h aP = loadA(&Pst[wv][n16][0], half);

    // --- O += P @ V (this wave's 192-col half)
#pragma unroll
    for (int ct = 0; ct < 12; ++ct) {
      int n = ch * 192 + ct * 16 + n16;
      v16h bv = loadB(&VsT[n][0], half * 16);
      O[ct] = wmma16(aP, bv, O[ct]);
    }
  }

  // finalize: O / l -> AT
#pragma unroll
  for (int ct = 0; ct < 12; ++ct) {
#pragma unroll
    for (int v = 0; v < 8; ++v) {
      size_t row = qrow0 + v + 8 * half;
      int c = ch * 192 + ct * 16 + n16;
      AT[row * (size_t)DIM + c] = (_Float16)(O[ct][v] / l_r[v]);
    }
  }
}

// ------- Kernel 4: project GEMM over concat(geglu_local, attention) + residual -------
__global__ __launch_bounds__(128) void k_project(const _Float16* __restrict__ G,
                                                 const _Float16* __restrict__ AT,
                                                 const float* __restrict__ pw,
                                                 const float* __restrict__ x,
                                                 float* __restrict__ out) {
  __shared__ __align__(16) _Float16 Xs[64][XST];
  __shared__ __align__(16) _Float16 Bs[64][XST];
  int row0 = blockIdx.x * 64;
  int c0   = blockIdx.y * 64;
  int tid  = threadIdx.x;
  int wv   = tid >> 5, lane = tid & 31, half = lane >> 4, n16 = lane & 15;

  v8f C[4];
#pragma unroll
  for (int t = 0; t < 4; ++t)
#pragma unroll
    for (int v = 0; v < 8; ++v) C[t][v] = 0.f;

  for (int k0 = 0; k0 < EXPD; k0 += 32) {
    __syncthreads();
#pragma unroll
    for (int i = 0; i < 2; ++i) {
      int idx = tid + 128 * i;
      int r = idx >> 2, part = idx & 3;
      const _Float16* src = (k0 < DIM)
          ? (G  + (size_t)(row0 + r) * EXPD + k0)
          : (AT + (size_t)(row0 + r) * DIM  + (k0 - DIM));
      *(uint4*)&Xs[r][part * 8] = *(const uint4*)(src + part * 8);
    }
#pragma unroll
    for (int i = 0; i < 4; ++i) {
      int idx = tid + 128 * i;
      int n = idx >> 3, kq = (idx & 7) * 4;
      float4 f = *(const float4*)(pw + (size_t)(c0 + n) * EXPD + k0 + kq);
      uint2 u;
      u.x = pack2h(f.x, f.y);
      u.y = pack2h(f.z, f.w);
      *(uint2*)&Bs[n][kq] = u;
    }
    __syncthreads();
    v16h a = loadA(&Xs[wv * 16 + n16][0], half);
#pragma unroll
    for (int nt = 0; nt < 4; ++nt) {
      v16h b = loadB(&Bs[nt * 16 + n16][0], half * 16);
      C[nt] = wmma16(a, b, C[nt]);
    }
  }
#pragma unroll
  for (int nt = 0; nt < 4; ++nt) {
#pragma unroll
    for (int v = 0; v < 8; ++v) {
      size_t row = (size_t)row0 + wv * 16 + v + 8 * half;
      int d = c0 + nt * 16 + n16;
      out[row * DIM + d] = x[row * DIM + d] + C[nt][v];
    }
  }
}

// ---------------- launcher ----------------
extern "C" void kernel_launch(void* const* d_in, const int* in_sizes, int n_in,
                              void* d_out, int out_size, void* d_ws, size_t ws_size,
                              hipStream_t stream) {
  const float* x  = (const float*)d_in[0];
  const float* nw = (const float*)d_in[1];
  const float* ew = (const float*)d_in[2];
  const float* pw = (const float*)d_in[3];
  const float* pm = (const float*)d_in[4];
  float* out = (float*)d_out;

  char* base = (char*)d_ws;
  _Float16* Xn = (_Float16*)base;  base += (size_t)ROWS * DIM * 2;   // 48 MB
  _Float16* Qb = (_Float16*)base;  base += (size_t)ROWS * 64 * 2;    // 8 MB
  _Float16* Kb = (_Float16*)base;  base += (size_t)ROWS * 64 * 2;    // 8 MB
  _Float16* G  = (_Float16*)base;  base += (size_t)ROWS * EXPD * 2;  // 96 MB
  _Float16* AT = (_Float16*)base;  base += (size_t)ROWS * DIM * 2;   // 48 MB
  (void)ws_size; (void)in_sizes; (void)n_in; (void)out_size;

  k_norm<<<ROWS / 8, 256, 0, stream>>>(x, nw, Xn);
  k_expand_qk<<<ROWS / 64, 128, 0, stream>>>(Xn, ew, Qb, Kb);
  k_expand_geglu<<<dim3(ROWS / 64, EXPD / 64), 128, 0, stream>>>(Xn, ew, G);
  k_attn<<<dim3(SS / 64, BB), 256, 0, stream>>>(Qb, Kb, G, pm, AT);
  k_project<<<dim3(ROWS / 64, DIM / 64), 128, 0, stream>>>(G, AT, pw, x, out);
}
